// E61DecayGated_4887672783108
// MI455X (gfx1250) — matrix operand order repaired and checked by reading, not compiled, not run on hardware
//
#include <hip/hip_runtime.h>

// Problem constants (match reference)
#define D_MODEL 1024
#define D_INNER 2048
#define BB      8
#define TT      2048
#define MROWS   (BB * TT)   // 16384

typedef __attribute__((ext_vector_type(16))) __bf16        bf16x16;
typedef __attribute__((ext_vector_type(8)))  float         f32x8;
typedef __attribute__((ext_vector_type(4)))  unsigned int  u32x4;

union Frag { u32x4 u[2]; bf16x16 v; };   // 32 bytes: one WMMA bf16 A/B fragment per lane

// round-to-nearest-even f32 -> bf16
__device__ __forceinline__ unsigned short f2bf(float f) {
  unsigned u = __float_as_uint(f);
  u += 0x7FFFu + ((u >> 16) & 1u);
  return (unsigned short)(u >> 16);
}

__global__ void cvt_f32_to_bf16(const float* __restrict__ in,
                                unsigned short* __restrict__ out, int n) {
  int i = blockIdx.x * blockDim.x + threadIdx.x;
  int stride = gridDim.x * blockDim.x;
  for (; i < n; i += stride) out[i] = f2bf(in[i]);
}

// Generic bf16 WMMA GEMM: C[M,N] = A[M,K] * Bw[N,K]^T, fp32 accumulate.
//   A row-major [M,K] bf16, Bw row-major [N,K] bf16 (i.e. B-matrix column n = W row n).
// Wave tile 32x64 (2 A-frags x 4 B-frags, 8 WMMAs / k-step). Block = 8 waves = 64x256.
// mode 0: silu(acc)           -> bf16, row remap b-major(b*T+t) -> t-major(t*B+b)
// mode 1: sigmoid(acc+bias)   -> f32, same row
// mode 2: acc+bias            -> f32, same row
// mode 3: acc                 -> f32, row remap t-major(t*B+b) -> b-major(b*T+t)
__global__ __launch_bounds__(256)
void gemm_wmma_bf16(const unsigned short* __restrict__ A,
                    const unsigned short* __restrict__ Bw,
                    const float* __restrict__ bias,
                    float* __restrict__ Cf,
                    unsigned short* __restrict__ Cb,
                    int M, int N, int K, int mode) {
  const int lane = threadIdx.x & 31;
  const int wave = threadIdx.x >> 5;
  const int half = lane >> 4;      // 0 or 1 (which half-wave)
  const int l16  = lane & 15;
  const int mtile = blockIdx.x * 64  + (wave >> 2) * 32;
  const int ntile = blockIdx.y * 256 + (wave & 3) * 64;

  f32x8 acc[2][4] = {};

  // A-fragment base pointers: row = mtile + mi*16 + l16; 16 contiguous K at half*16
  const unsigned short* a0 = A + (size_t)(mtile + l16) * K + half * 16;
  const unsigned short* a1 = a0 + (size_t)16 * K;
  // B-fragment base pointers: col n = ntile + ni*16 + l16; 16 contiguous K at half*16
  const unsigned short* bp0 = Bw + (size_t)(ntile +  0 + l16) * K + half * 16;
  const unsigned short* bp1 = Bw + (size_t)(ntile + 16 + l16) * K + half * 16;
  const unsigned short* bp2 = Bw + (size_t)(ntile + 32 + l16) * K + half * 16;
  const unsigned short* bp3 = Bw + (size_t)(ntile + 48 + l16) * K + half * 16;

  for (int kk = 0; kk < K; kk += 32) {
    Frag fa0, fa1, fb0, fb1, fb2, fb3;
    fa0.u[0] = *(const u32x4*)(a0 + kk);       // K = kk+half*16 .. +8
    fa0.u[1] = *(const u32x4*)(a0 + kk + 8);   // K = kk+half*16+8 .. +8
    fa1.u[0] = *(const u32x4*)(a1 + kk);
    fa1.u[1] = *(const u32x4*)(a1 + kk + 8);
    fb0.u[0] = *(const u32x4*)(bp0 + kk); fb0.u[1] = *(const u32x4*)(bp0 + kk + 8);
    fb1.u[0] = *(const u32x4*)(bp1 + kk); fb1.u[1] = *(const u32x4*)(bp1 + kk + 8);
    fb2.u[0] = *(const u32x4*)(bp2 + kk); fb2.u[1] = *(const u32x4*)(bp2 + kk + 8);
    fb3.u[0] = *(const u32x4*)(bp3 + kk); fb3.u[1] = *(const u32x4*)(bp3 + kk + 8);

    acc[0][0] = __builtin_amdgcn_wmma_f32_16x16x32_bf16(false, fa0.v, false, fb0.v, (short)0, acc[0][0], false, false);
    acc[0][1] = __builtin_amdgcn_wmma_f32_16x16x32_bf16(false, fa0.v, false, fb1.v, (short)0, acc[0][1], false, false);
    acc[0][2] = __builtin_amdgcn_wmma_f32_16x16x32_bf16(false, fa0.v, false, fb2.v, (short)0, acc[0][2], false, false);
    acc[0][3] = __builtin_amdgcn_wmma_f32_16x16x32_bf16(false, fa0.v, false, fb3.v, (short)0, acc[0][3], false, false);
    acc[1][0] = __builtin_amdgcn_wmma_f32_16x16x32_bf16(false, fa1.v, false, fb0.v, (short)0, acc[1][0], false, false);
    acc[1][1] = __builtin_amdgcn_wmma_f32_16x16x32_bf16(false, fa1.v, false, fb1.v, (short)0, acc[1][1], false, false);
    acc[1][2] = __builtin_amdgcn_wmma_f32_16x16x32_bf16(false, fa1.v, false, fb2.v, (short)0, acc[1][2], false, false);
    acc[1][3] = __builtin_amdgcn_wmma_f32_16x16x32_bf16(false, fa1.v, false, fb3.v, (short)0, acc[1][3], false, false);
  }

  // Epilogue. C layout: VGPR j holds M = j (lanes 0-15) / j+8 (lanes 16-31), N = l16.
  #pragma unroll
  for (int mi = 0; mi < 2; ++mi) {
    #pragma unroll
    for (int j = 0; j < 8; ++j) {
      const int row = mtile + mi * 16 + half * 8 + j;
      #pragma unroll
      for (int ni = 0; ni < 4; ++ni) {
        const int col = ntile + ni * 16 + l16;
        const float x = acc[mi][ni][j];
        if (mode == 0) {
          // silu, b-major row (b*T+t) -> t-major (t*B+b), bf16 output
          const float s = x / (1.f + __expf(-x));
          const int b = row / TT, t = row % TT;
          Cb[(size_t)(t * BB + b) * N + col] = f2bf(s);
        } else if (mode == 1) {
          const float y = x + bias[col];
          Cf[(size_t)row * N + col] = 1.f / (1.f + __expf(-y));
        } else if (mode == 2) {
          Cf[(size_t)row * N + col] = x + bias[col];
        } else {
          // t-major row (t*B+b) -> b-major (b*T+t), f32 output
          const int t = row / BB, b = row % BB;
          Cf[(size_t)(b * TT + t) * N + col] = x;
        }
      }
    }
  }
}

// Sequential decay-gated scan, fp32 recurrence (exact), bf16 outs for final GEMM.
// 16384 threads: thread i = (b = i/2048, e = i%2048); consecutive threads -> contiguous e.
__global__ __launch_bounds__(256)
void scan_kernel(const float* __restrict__ alpha, const float* __restrict__ vv,
                 const float* __restrict__ h0,
                 unsigned short* __restrict__ outs_bf, float* __restrict__ hfin) {
  const int i = blockIdx.x * blockDim.x + threadIdx.x;  // 0..16383
  float h = h0[i];
  size_t idx = (size_t)i;  // (t=0, b, e) index == b*D_INNER + e == i
  for (int t = 0; t < TT; ++t) {
    const float a  = alpha[idx];
    const float vt = vv[idx];
    h = a * h + (1.f - a) * vt;
    const float sg = 1.f / (1.f + __expf(-h));
    outs_bf[idx] = f2bf(h * h * sg);   // h * silu(h)
    idx += (size_t)BB * D_INNER;
  }
  hfin[i] = h;
}

extern "C" void kernel_launch(void* const* d_in, const int* in_sizes, int n_in,
                              void* d_out, int out_size, void* d_ws, size_t ws_size,
                              hipStream_t stream) {
  (void)in_sizes; (void)n_in; (void)out_size; (void)ws_size;
  const float* x       = (const float*)d_in[0];  // [B,T,D_MODEL]
  const float* h0      = (const float*)d_in[1];  // [B,D_INNER]
  const float* W_in    = (const float*)d_in[2];  // [D_INNER,D_MODEL]
  const float* W_alpha = (const float*)d_in[3];  // [D_INNER,D_INNER]
  const float* b_alpha = (const float*)d_in[4];  // [D_INNER]
  const float* W_v     = (const float*)d_in[5];  // [D_INNER,D_INNER]
  const float* b_v     = (const float*)d_in[6];  // [D_INNER]
  const float* W_out   = (const float*)d_in[7];  // [D_MODEL,D_INNER]

  float* out  = (float*)d_out;                          // [B,T,D_MODEL]
  float* hfin = out + (size_t)BB * TT * D_MODEL;        // [B,D_INNER] tail

  // Workspace carve-out (~440 MB total)
  char* ws = (char*)d_ws;
  size_t off = 0;
  auto carve = [&](size_t bytes) -> char* {
    char* p = ws + off;
    off += (bytes + 255) & ~(size_t)255;
    return p;
  };
  unsigned short* x_bf     = (unsigned short*)carve((size_t)MROWS * D_MODEL * 2);
  unsigned short* Win_bf   = (unsigned short*)carve((size_t)D_INNER * D_MODEL * 2);
  unsigned short* Wal_bf   = (unsigned short*)carve((size_t)D_INNER * D_INNER * 2);
  unsigned short* Wv_bf    = (unsigned short*)carve((size_t)D_INNER * D_INNER * 2);
  unsigned short* Wout_bf  = (unsigned short*)carve((size_t)D_MODEL * D_INNER * 2);
  unsigned short* xproj_bf = (unsigned short*)carve((size_t)MROWS * D_INNER * 2);  // t-major
  float*          alpha_f  = (float*)carve((size_t)MROWS * D_INNER * 4);           // t-major
  float*          v_f      = (float*)carve((size_t)MROWS * D_INNER * 4);           // t-major
  unsigned short* outs_bf  = (unsigned short*)carve((size_t)MROWS * D_INNER * 2);  // t-major

  // 1) f32 -> bf16 conversions
  {
    const int nx = MROWS * D_MODEL;
    cvt_f32_to_bf16<<<1024, 256, 0, stream>>>(x, x_bf, nx);
    cvt_f32_to_bf16<<<512,  256, 0, stream>>>(W_in, Win_bf, D_INNER * D_MODEL);
    cvt_f32_to_bf16<<<512,  256, 0, stream>>>(W_alpha, Wal_bf, D_INNER * D_INNER);
    cvt_f32_to_bf16<<<512,  256, 0, stream>>>(W_v, Wv_bf, D_INNER * D_INNER);
    cvt_f32_to_bf16<<<512,  256, 0, stream>>>(W_out, Wout_bf, D_MODEL * D_INNER);
  }

  const dim3 blk(256);
  // 2) x_proj = silu(x @ W_in^T), written t-major bf16  [M=16384, N=2048, K=1024]
  gemm_wmma_bf16<<<dim3(MROWS / 64, D_INNER / 256), blk, 0, stream>>>(
      x_bf, Win_bf, nullptr, nullptr, xproj_bf, MROWS, D_INNER, D_MODEL, 0);
  // 3) alpha = sigmoid(x_proj @ W_alpha^T + b_alpha), f32  [16384, 2048, 2048]
  gemm_wmma_bf16<<<dim3(MROWS / 64, D_INNER / 256), blk, 0, stream>>>(
      xproj_bf, Wal_bf, b_alpha, alpha_f, nullptr, MROWS, D_INNER, D_INNER, 1);
  //    v = x_proj @ W_v^T + b_v, f32
  gemm_wmma_bf16<<<dim3(MROWS / 64, D_INNER / 256), blk, 0, stream>>>(
      xproj_bf, Wv_bf, b_v, v_f, nullptr, MROWS, D_INNER, D_INNER, 2);
  // 4) sequential scan (fp32), outs -> bf16, h_final -> output tail
  scan_kernel<<<dim3((BB * D_INNER) / 256), blk, 0, stream>>>(
      alpha_f, v_f, h0, outs_bf, hfin);
  // 5) output = outs @ W_out^T, remap to [B,T,D_MODEL] f32  [16384, 1024, 2048]
  gemm_wmma_bf16<<<dim3(MROWS / 64, D_MODEL / 256), blk, 0, stream>>>(
      outs_bf, Wout_bf, nullptr, out, nullptr, MROWS, D_MODEL, D_INNER, 3);
}
